// Mamba_38878043963458
// MI455X (gfx1250) — matrix operand order repaired and checked by reading, not compiled, not run on hardware
//
#include <hip/hip_runtime.h>
#include <stdint.h>

#define EPSV 1e-5f
#define D_MODEL 512
#define D_INNER 1024
#define D_STATE 16
#define DT_RANK 32
#define D_CONV 4
#define DEPTH 24
#define BATCH 8
#define L_SEQ 256
#define MROWS (BATCH * L_SEQ) /* 2048 */
#define XDBL_N (DT_RANK + 2 * D_STATE) /* 64 */

// GEMM block tiling: 128x64 block tile, 8 waves of 32x32 register tiles, K-step 32
#define BM 128
#define BN 64
#define BK 32

typedef __bf16 bf16_t;
typedef __attribute__((ext_vector_type(16))) __bf16 v16bf;
typedef __attribute__((ext_vector_type(8)))  __bf16 v8bf;
typedef __attribute__((ext_vector_type(8)))  float  v8f;
typedef __attribute__((ext_vector_type(4)))  unsigned int u32x4;
typedef __attribute__((ext_vector_type(4)))  int i32x4;
typedef __attribute__((ext_vector_type(8)))  int i32x8;

__device__ __forceinline__ float siluf(float x) { return x / (1.f + __expf(-x)); }
__device__ __forceinline__ float softplusf(float x) {
  return (x > 20.f) ? x : logf(1.f + __expf(x));
}

// Flat->LDS: for LDS-aperture addresses the low 32 bits are the LDS byte offset.
__device__ __forceinline__ unsigned int lds_byte_off(const void* p) {
  return (unsigned int)(uintptr_t)p;
}

// ---------------------------------------------------------------------------
// TDM: issue a 2D tile DMA (global row-major [rows x K] slab -> LDS, packed
// row-major tile_d1 x tile_d0 elements of 2 bytes). D# per CDNA5 ISA §8.3/8.4.
//   group0: count=1 | lds_addr | global_addr[56:0] | type=2
//   group1: data_size=1(2B), tensor_dim0=K, tensor_dim1=rows,
//           tile_dim0, tile_dim1, tensor_dim0_stride=K (element units)
// ---------------------------------------------------------------------------
__device__ __forceinline__ void tdm_load_2d(const bf16_t* g, unsigned int ldsOff,
                                            unsigned int tile_d0, unsigned int tile_d1,
                                            unsigned int t_d0, unsigned int t_d1,
                                            unsigned long long stride0) {
  unsigned long long ga = (unsigned long long)(uintptr_t)g;
  u32x4 g0;
  g0[0] = 1u;                                            // count=1 (valid user D#)
  g0[1] = ldsOff;                                        // lds_addr (bytes)
  g0[2] = (unsigned int)ga;                              // global_addr[31:0]
  g0[3] = (unsigned int)((ga >> 32) & 0x1FFFFFFu) | (2u << 30); // addr[56:32] | type=2
  i32x8 g1;
  g1[0] = (int)(1u << 16);                               // data_size=1 (2 bytes), mask=0
  g1[1] = (int)((t_d0 & 0xFFFFu) << 16);                 // tensor_dim0[15:0] at bits 63:48
  g1[2] = (int)((t_d0 >> 16) | ((t_d1 & 0xFFFFu) << 16));// tensor_dim0[31:16] | tensor_dim1[15:0]
  g1[3] = (int)(((t_d1 >> 16) & 0xFFFFu) | (tile_d0 << 16)); // tensor_dim1[31:16] | tile_dim0
  g1[4] = (int)(tile_d1 & 0xFFFFu);                      // tile_dim1, tile_dim2=0
  g1[5] = (int)(stride0 & 0xFFFFFFFFu);                  // tensor_dim0_stride[31:0]
  g1[6] = (int)((stride0 >> 32) & 0xFFFFu);              // stride0[47:32] | stride1[15:0]=0
  g1[7] = 0;
  i32x4 z4 = {0, 0, 0, 0};
#if defined(__clang_major__) && (__clang_major__ >= 23)
  i32x8 z8 = {0, 0, 0, 0, 0, 0, 0, 0};
  __builtin_amdgcn_tensor_load_to_lds(g0, g1, z4, z4, z8, 0);
#else
  __builtin_amdgcn_tensor_load_to_lds(g0, g1, z4, z4, 0);
#endif
}

// ---------------------------------------------------------------------------
// f32 -> bf16 conversion
// ---------------------------------------------------------------------------
__global__ void k_f32_to_bf16(const float* __restrict__ in, bf16_t* __restrict__ out, int n) {
  int i = blockIdx.x * blockDim.x + threadIdx.x;
  if (i < n) out[i] = (bf16_t)in[i];
}

// ---------------------------------------------------------------------------
// resid = x + resid ; h = rmsnorm(resid, w)  -> bf16 and/or f32 outputs
// ---------------------------------------------------------------------------
__global__ __launch_bounds__(256)
void k_add_rmsnorm(const float* __restrict__ x, float* __restrict__ resid,
                   const float* __restrict__ w,
                   bf16_t* __restrict__ out_bf, float* __restrict__ out_f32) {
  __shared__ float red[256];
  int row = blockIdx.x;
  int t = threadIdx.x;
  size_t base = (size_t)row * D_MODEL;
  float v0 = x[base + t]       + resid[base + t];
  float v1 = x[base + t + 256] + resid[base + t + 256];
  resid[base + t]       = v0;
  resid[base + t + 256] = v1;
  red[t] = v0 * v0 + v1 * v1;
  __syncthreads();
  for (int s = 128; s > 0; s >>= 1) {
    if (t < s) red[t] += red[t + s];
    __syncthreads();
  }
  float rs = rsqrtf(red[0] / (float)D_MODEL + EPSV);
  float h0 = v0 * rs * w[t];
  float h1 = v1 * rs * w[t + 256];
  if (out_bf)  { out_bf[base + t] = (bf16_t)h0;  out_bf[base + t + 256] = (bf16_t)h1; }
  if (out_f32) { out_f32[base + t] = h0;         out_f32[base + t + 256] = h1; }
}

// ---------------------------------------------------------------------------
// Blocked WMMA bf16 GEMM with TDM double-buffered LDS staging.
//   C[M,N] = A[M,K] @ W[N,K]^T   (f32 accumulate)
// Block: 128(M) x 64(N); 8 waves in 4x2 grid; wave tile 32x32 (2x2 WMMA tiles).
// Per K-step: wave0 issues 2 TDM tile DMAs for the next K-slab while all waves
// compute from the current LDS buffers; sync = s_wait_tensorcnt + barrier.
// Fragment layouts used here, same k order for A and B (A: row/lane, K split by half;
// B: col/lane, K = 16*half + e); C/D: m = 8*half + j, n = lane&15.
// epi: 0 = plain, 1 = softplus(acc + bias[n]), 2 = 0.5*acc
// ---------------------------------------------------------------------------
__global__ __launch_bounds__(256)
void k_wmma_gemm(const bf16_t* __restrict__ A, const bf16_t* __restrict__ W,
                 const float* __restrict__ bias, float* __restrict__ C,
                 int M, int N, int K, int epi) {
  __shared__ __align__(16) bf16_t lds[2 * (BM * BK + BN * BK)];

  int nblocks = N / BN;
  int bm = blockIdx.x / nblocks, bn = blockIdx.x % nblocks;
  int wave = threadIdx.x >> 5, lane = threadIdx.x & 31;
  int half = lane >> 4, r = lane & 15;
  int wm = wave >> 1, wn = wave & 1;

  const bf16_t* Ablk = A + (size_t)bm * BM * K;
  const bf16_t* Wblk = W + (size_t)bn * BN * K;

  unsigned int ldsBase = lds_byte_off(lds);
  const unsigned int bufBytes = (BM * BK + BN * BK) * 2;

  v8f acc[2][2] = {};

  int ksteps = K / BK;
  if (wave == 0) {
    tdm_load_2d(Ablk, ldsBase, BK, BM, (unsigned)K, BM, (unsigned long long)K);
    tdm_load_2d(Wblk, ldsBase + BM * BK * 2, BK, BN, (unsigned)K, BN, (unsigned long long)K);
    __builtin_amdgcn_s_wait_tensorcnt(0);
  }
  __syncthreads();

  for (int ks = 0; ks < ksteps; ks++) {
    int cur = ks & 1;
    if (wave == 0 && (ks + 1) < ksteps) {
      int nxt = cur ^ 1;
      const bf16_t* ga = Ablk + (size_t)(ks + 1) * BK;
      const bf16_t* gb = Wblk + (size_t)(ks + 1) * BK;
      tdm_load_2d(ga, ldsBase + nxt * bufBytes, BK, BM, (unsigned)K, BM, (unsigned long long)K);
      tdm_load_2d(gb, ldsBase + nxt * bufBytes + BM * BK * 2, BK, BN, (unsigned)K, BN,
                  (unsigned long long)K);
    }

    const bf16_t* Ab = lds + cur * (BM * BK + BN * BK);
    const bf16_t* Bb = Ab + BM * BK;

    v16bf af[2], bfr[2];
#pragma unroll
    for (int i = 0; i < 2; i++) {
      int row = wm * 32 + i * 16 + r;
      v8bf lo = *(const v8bf*)(Ab + row * BK + 16 * half);
      v8bf hi = *(const v8bf*)(Ab + row * BK + 16 * half + 8);
#pragma unroll
      for (int q = 0; q < 8; q++) { af[i][q] = lo[q]; af[i][q + 8] = hi[q]; }
    }
#pragma unroll
    for (int j = 0; j < 2; j++) {
      int row = wn * 32 + j * 16 + r;
      v8bf lo = *(const v8bf*)(Bb + row * BK + 16 * half);
      v8bf hi = *(const v8bf*)(Bb + row * BK + 16 * half + 8);
#pragma unroll
      for (int q = 0; q < 8; q++) { bfr[j][q] = lo[q]; bfr[j][q + 8] = hi[q]; }
    }

#pragma unroll
    for (int i = 0; i < 2; i++)
#pragma unroll
      for (int j = 0; j < 2; j++)
        acc[i][j] = __builtin_amdgcn_wmma_f32_16x16x32_bf16(
            false, af[i], false, bfr[j], (short)0, acc[i][j], false, false);

    if (wave == 0) __builtin_amdgcn_s_wait_tensorcnt(0);
    __syncthreads();
  }

#pragma unroll
  for (int i = 0; i < 2; i++) {
#pragma unroll
    for (int j = 0; j < 2; j++) {
      int n = bn * BN + wn * 32 + j * 16 + r;
      float bval = (epi == 1) ? bias[n] : 0.f;
#pragma unroll
      for (int q = 0; q < 8; q++) {
        int m = bm * BM + wm * 32 + i * 16 + 8 * half + q;
        float v = acc[i][j][q];
        if (epi == 1)      v = softplusf(v + bval);
        else if (epi == 2) v *= 0.5f;
        C[(size_t)m * N + n] = v;
      }
    }
  }
}

// ---------------------------------------------------------------------------
// causal depthwise conv1d (taps=4) + SiLU over the x-half of xz.
// ---------------------------------------------------------------------------
__global__ void k_conv1d_silu(const float* __restrict__ xz, const float* __restrict__ cw,
                              const float* __restrict__ cb, float* __restrict__ xc,
                              bf16_t* __restrict__ xc_bf, int flip) {
  int i = blockIdx.x * blockDim.x + threadIdx.x;
  if (i >= MROWS * D_INNER) return;
  int d = i % D_INNER;
  int m = i / D_INNER;
  int b = m / L_SEQ, l = m % L_SEQ;
  float acc = cb[d];
#pragma unroll
  for (int k = 0; k < D_CONV; k++) {
    int lp = l + k - (D_CONV - 1);
    if (lp >= 0) {
      int lsrc = flip ? (L_SEQ - 1 - lp) : lp;
      acc += xz[(size_t)(b * L_SEQ + lsrc) * (2 * D_INNER) + d] * cw[d * D_CONV + k];
    }
  }
  float v = siluf(acc);
  xc[(size_t)m * D_INNER + d]    = v;
  xc_bf[(size_t)m * D_INNER + d] = (bf16_t)v;
}

// dt_lo slice (first 32 cols of xdbl) -> bf16 [M, 32]
__global__ void k_slice_dtlo(const float* __restrict__ xdbl, bf16_t* __restrict__ out) {
  int i = blockIdx.x * blockDim.x + threadIdx.x;
  if (i >= MROWS * DT_RANK) return;
  int m = i / DT_RANK, r = i % DT_RANK;
  out[i] = (bf16_t)xdbl[(size_t)m * XDBL_N + r];
}

// ---------------------------------------------------------------------------
// selective scan: one thread per (b,d); 16 states in registers, L=256 steps.
// ---------------------------------------------------------------------------
__global__ __launch_bounds__(256)
void k_scan(const float* __restrict__ dt, const float* __restrict__ xc,
            const float* __restrict__ xdbl, const float* __restrict__ xz,
            const float* __restrict__ A_log, const float* __restrict__ Dp,
            float* __restrict__ ysum, int flip, int accum) {
  int i = blockIdx.x * blockDim.x + threadIdx.x;
  if (i >= BATCH * D_INNER) return;
  int b = i / D_INNER, d = i % D_INNER;
  float A[D_STATE], h[D_STATE];
#pragma unroll
  for (int n = 0; n < D_STATE; n++) {
    A[n] = -__expf(A_log[d * D_STATE + n]);
    h[n] = 0.f;
  }
  float Dd = Dp[d];
  for (int l = 0; l < L_SEQ; l++) {
    size_t row = (size_t)(b * L_SEQ + l);
    float dtv = dt[row * D_INNER + d];
    float u   = xc[row * D_INNER + d];
    float du = dtv * u;
    const float* Bc = xdbl + row * XDBL_N + DT_RANK;
    const float* Cc = Bc + D_STATE;
    float y = 0.f;
#pragma unroll
    for (int n = 0; n < D_STATE; n++) {
      h[n] = __expf(dtv * A[n]) * h[n] + du * Bc[n];
      y = fmaf(h[n], Cc[n], y);
    }
    y = fmaf(u, Dd, y);
    int lo = flip ? (L_SEQ - 1 - l) : l;
    size_t orow = (size_t)(b * L_SEQ + lo);
    float z = xz[orow * (2 * D_INNER) + D_INNER + d];
    y *= siluf(z);
    size_t oi = orow * D_INNER + d;
    if (accum) ysum[oi] += y; else ysum[oi] = y;
  }
}

// ---------------------------------------------------------------------------
// patch embed: 16x16/16 conv + bias + pos_embed; also zeroes resid.
// ---------------------------------------------------------------------------
__global__ void k_patch_embed(const float* __restrict__ x, const float* __restrict__ pw,
                              const float* __restrict__ pb, const float* __restrict__ pos,
                              float* __restrict__ cur, float* __restrict__ resid) {
  int i = blockIdx.x * blockDim.x + threadIdx.x;
  if (i >= MROWS * D_MODEL) return;
  int c = i % D_MODEL;
  int m = i / D_MODEL;
  int b = m / L_SEQ, l = m % L_SEQ;
  int gy = l >> 4, gx = l & 15;
  float acc = pb[c];
  for (int ci = 0; ci < 3; ci++) {
    const float* xb = x + ((size_t)(b * 3 + ci) * 256 + gy * 16) * 256 + gx * 16;
    const float* wb = pw + (size_t)(c * 3 + ci) * 256;
    for (int ky = 0; ky < 16; ky++)
#pragma unroll 4
      for (int kx = 0; kx < 16; kx++)
        acc = fmaf(xb[ky * 256 + kx], wb[ky * 16 + kx], acc);
  }
  cur[i] = acc + pos[(size_t)l * D_MODEL + c];
  resid[i] = 0.f;
}

// [B,L,C] -> [B,C,16,16]
__global__ void k_seq_to_chw(const float* __restrict__ h, float* __restrict__ out) {
  int i = blockIdx.x * blockDim.x + threadIdx.x;
  if (i >= BATCH * D_MODEL * L_SEQ) return;
  int l = i % L_SEQ; int t = i / L_SEQ; int c = t % D_MODEL; int b = t / D_MODEL;
  out[i] = h[(size_t)(b * L_SEQ + l) * D_MODEL + c];
}

// 3x3 VALID conv, (B,512,H,H) -> (B,512,H-2,H-2)
__global__ void k_conv3x3(const float* __restrict__ in, const float* __restrict__ w,
                          const float* __restrict__ bias, float* __restrict__ out, int H) {
  int HO = H - 2;
  int i = blockIdx.x * blockDim.x + threadIdx.x;
  if (i >= BATCH * D_MODEL * HO * HO) return;
  int x = i % HO; int t = i / HO;
  int y = t % HO; t /= HO;
  int co = t % D_MODEL; int b = t / D_MODEL;
  float acc = bias[co];
  for (int ci = 0; ci < D_MODEL; ci++) {
    const float* ib = in + ((size_t)(b * D_MODEL + ci) * H + y) * H + x;
    const float* wb = w + (size_t)(co * D_MODEL + ci) * 9;
#pragma unroll
    for (int ky = 0; ky < 3; ky++)
#pragma unroll
      for (int kx = 0; kx < 3; kx++)
        acc = fmaf(ib[ky * H + kx], wb[ky * 3 + kx], acc);
  }
  out[i] = acc;
}

// adaptive average pool (n_in -> n_out per dim), (B,512,n,n) -> (B,512,m,m)
__global__ void k_adaptive_pool(const float* __restrict__ in, float* __restrict__ out,
                                int n_in, int n_out) {
  int i = blockIdx.x * blockDim.x + threadIdx.x;
  if (i >= BATCH * D_MODEL * n_out * n_out) return;
  int q = i % n_out; int t = i / n_out;
  int p = t % n_out; t /= n_out;
  int c = t % D_MODEL; int b = t / D_MODEL;
  int sy = p * n_in / n_out, ey = ((p + 1) * n_in + n_out - 1) / n_out;
  int sx = q * n_in / n_out, ex = ((q + 1) * n_in + n_out - 1) / n_out;
  const float* ib = in + (size_t)(b * D_MODEL + c) * n_in * n_in;
  float acc = 0.f;
  for (int y = sy; y < ey; y++)
    for (int x = sx; x < ex; x++)
      acc += ib[y * n_in + x];
  out[i] = acc * (1.f / (float)(ey - sy)) * (1.f / (float)(ex - sx));
}

// (B,512,16) -> (B,16,512)
__global__ void k_final_out(const float* __restrict__ zz, float* __restrict__ out) {
  int i = blockIdx.x * blockDim.x + threadIdx.x;
  if (i >= BATCH * 16 * D_MODEL) return;
  int c = i % D_MODEL;
  int t = i / D_MODEL;
  int b = t / 16, j = t % 16;
  out[i] = zz[(size_t)(b * D_MODEL + c) * 16 + j];
}

// ---------------------------------------------------------------------------
static inline int cdiv(long long a, long long b) { return (int)((a + b - 1) / b); }

extern "C" void kernel_launch(void* const* d_in, const int* in_sizes, int n_in,
                              void* d_out, int out_size, void* d_ws, size_t ws_size,
                              hipStream_t stream) {
  (void)in_sizes; (void)n_in; (void)out_size; (void)ws_size;
  const float* x       = (const float*)d_in[0];
  const float* patch_w = (const float*)d_in[1];
  const float* patch_b = (const float*)d_in[2];
  const float* pos     = (const float*)d_in[3];
  const float* norm_w  = (const float*)d_in[4];
  const float* in_w    = (const float*)d_in[5];
  const float* cw_f    = (const float*)d_in[6];
  const float* cb_f    = (const float*)d_in[7];
  const float* xp_f    = (const float*)d_in[8];
  const float* dtw_f   = (const float*)d_in[9];
  const float* dtb_f   = (const float*)d_in[10];
  const float* Alog_f  = (const float*)d_in[11];
  const float* Dp_f    = (const float*)d_in[12];
  const float* cw_b    = (const float*)d_in[13];
  const float* cb_b    = (const float*)d_in[14];
  const float* xp_b    = (const float*)d_in[15];
  const float* dtw_b   = (const float*)d_in[16];
  const float* dtb_b   = (const float*)d_in[17];
  const float* Alog_b  = (const float*)d_in[18];
  const float* Dp_b    = (const float*)d_in[19];
  const float* out_w   = (const float*)d_in[20];
  const float* norm_fw = (const float*)d_in[21];
  const float* c1w     = (const float*)d_in[22];
  const float* c1b     = (const float*)d_in[23];
  const float* c2w     = (const float*)d_in[24];
  const float* c2b     = (const float*)d_in[25];
  float* out = (float*)d_out;

  // ---- workspace carve-out -------------------------------------------------
  char* p = (char*)d_ws;
  auto alloc = [&](size_t bytes) -> char* {
    char* r = p;
    p += (bytes + 255) & ~(size_t)255;
    return r;
  };
  float*  cur      = (float*)alloc((size_t)MROWS * D_MODEL * 4);
  float*  resid    = (float*)alloc((size_t)MROWS * D_MODEL * 4);
  bf16_t* hn_bf    = (bf16_t*)alloc((size_t)MROWS * D_MODEL * 2);
  float*  xz       = (float*)alloc((size_t)MROWS * 2 * D_INNER * 4);
  float*  xc       = (float*)alloc((size_t)MROWS * D_INNER * 4);
  bf16_t* xc_bf    = (bf16_t*)alloc((size_t)MROWS * D_INNER * 2);
  float*  xdbl     = (float*)alloc((size_t)MROWS * XDBL_N * 4);
  bf16_t* dtlo_bf  = (bf16_t*)alloc((size_t)MROWS * DT_RANK * 2);
  float*  dtbuf    = (float*)alloc((size_t)MROWS * D_INNER * 4);
  float*  ysum     = (float*)alloc((size_t)MROWS * D_INNER * 4);
  bf16_t* ysum_bf  = (bf16_t*)alloc((size_t)MROWS * D_INNER * 2);
  bf16_t* w_in_bf  = (bf16_t*)alloc((size_t)2 * D_INNER * D_MODEL * 2);
  bf16_t* w_xp_bf  = (bf16_t*)alloc((size_t)XDBL_N * D_INNER * 2);
  bf16_t* w_dt_bf  = (bf16_t*)alloc((size_t)D_INNER * DT_RANK * 2);
  bf16_t* w_out_bf = (bf16_t*)alloc((size_t)D_MODEL * D_INNER * 2);
  float*  hfin     = (float*)alloc((size_t)MROWS * D_MODEL * 4);
  float*  head_t   = (float*)alloc((size_t)BATCH * D_MODEL * 256 * 4);
  float*  head_a   = (float*)alloc((size_t)BATCH * D_MODEL * 196 * 4);
  float*  head_b   = (float*)alloc((size_t)BATCH * D_MODEL * 64 * 4);

  auto cvt = [&](const float* src, bf16_t* dst, int n) {
    k_f32_to_bf16<<<cdiv(n, 256), 256, 0, stream>>>(src, dst, n);
  };
  auto gemm = [&](const bf16_t* A, const bf16_t* W, const float* bias, float* C,
                  int M, int N, int K, int epi) {
    int blocks = (M / BM) * (N / BN);
    k_wmma_gemm<<<blocks, 256, 0, stream>>>(A, W, bias, C, M, N, K, epi);
  };

  // ---- patch embed + pos_embed, resid = 0 ---------------------------------
  k_patch_embed<<<cdiv((size_t)MROWS * D_MODEL, 256), 256, 0, stream>>>(
      x, patch_w, patch_b, pos, cur, resid);

  // ---- 24 bidirectional Mamba blocks --------------------------------------
  for (int lyr = 0; lyr < DEPTH; lyr++) {
    const float* nw = norm_w + (size_t)lyr * D_MODEL;
    const float* iw = in_w   + (size_t)lyr * 2 * D_INNER * D_MODEL;
    const float* ow = out_w  + (size_t)lyr * D_MODEL * D_INNER;

    // resid += x ; hn = rmsnorm(resid) (bf16 for GEMM)
    k_add_rmsnorm<<<MROWS, 256, 0, stream>>>(cur, resid, nw, hn_bf, (float*)nullptr);

    // in_proj: xz[M, 2048] = hn @ iw^T
    cvt(iw, w_in_bf, 2 * D_INNER * D_MODEL);
    gemm(hn_bf, w_in_bf, nullptr, xz, MROWS, 2 * D_INNER, D_MODEL, 0);

    for (int dir = 0; dir < 2; dir++) {
      const float* cw   = dir ? cw_b   : cw_f;
      const float* cb   = dir ? cb_b   : cb_f;
      const float* xp   = dir ? xp_b   : xp_f;
      const float* dtw  = dir ? dtw_b  : dtw_f;
      const float* dtb  = dir ? dtb_b  : dtb_f;
      const float* Alog = dir ? Alog_b : Alog_f;
      const float* Dp   = dir ? Dp_b   : Dp_f;
      const float* cwl  = cw   + (size_t)lyr * D_INNER * D_CONV;
      const float* cbl  = cb   + (size_t)lyr * D_INNER;
      const float* xpl  = xp   + (size_t)lyr * XDBL_N * D_INNER;
      const float* dtwl = dtw  + (size_t)lyr * D_INNER * DT_RANK;
      const float* dtbl = dtb  + (size_t)lyr * D_INNER;
      const float* All  = Alog + (size_t)lyr * D_INNER * D_STATE;
      const float* Dpl  = Dp   + (size_t)lyr * D_INNER;

      // depthwise causal conv + silu (flipped coords for dir==1)
      k_conv1d_silu<<<cdiv((size_t)MROWS * D_INNER, 256), 256, 0, stream>>>(
          xz, cwl, cbl, xc, xc_bf, dir);

      // x_proj: xdbl[M, 64] = xc @ xpl^T
      cvt(xpl, w_xp_bf, XDBL_N * D_INNER);
      gemm(xc_bf, w_xp_bf, nullptr, xdbl, MROWS, XDBL_N, D_INNER, 0);

      // dt = softplus(dt_lo @ dtw^T + dtb)
      k_slice_dtlo<<<cdiv((size_t)MROWS * DT_RANK, 256), 256, 0, stream>>>(xdbl, dtlo_bf);
      cvt(dtwl, w_dt_bf, D_INNER * DT_RANK);
      gemm(dtlo_bf, w_dt_bf, dtbl, dtbuf, MROWS, D_INNER, DT_RANK, 1);

      // selective scan + silu(z) gate; dir==0 stores, dir==1 accumulates
      k_scan<<<cdiv(BATCH * D_INNER, 256), 256, 0, stream>>>(
          dtbuf, xc, xdbl, xz, All, Dpl, ysum, dir, dir);
    }

    // out_proj: cur[M, 512] = 0.5 * (y_f + y_b) @ ow^T
    cvt(ysum, ysum_bf, MROWS * D_INNER);
    cvt(ow, w_out_bf, D_MODEL * D_INNER);
    gemm(ysum_bf, w_out_bf, nullptr, cur, MROWS, D_MODEL, D_INNER, 2);
  }

  // ---- final rmsnorm ------------------------------------------------------
  k_add_rmsnorm<<<MROWS, 256, 0, stream>>>(cur, resid, norm_fw, (bf16_t*)nullptr, hfin);

  // ---- head: conv3x3 -> pool(14->8) -> conv3x3 -> pool(6->4) --------------
  k_seq_to_chw<<<cdiv(BATCH * D_MODEL * L_SEQ, 256), 256, 0, stream>>>(hfin, head_t);
  k_conv3x3<<<cdiv(BATCH * D_MODEL * 14 * 14, 256), 256, 0, stream>>>(head_t, c1w, c1b, head_a, 16);
  k_adaptive_pool<<<cdiv(BATCH * D_MODEL * 8 * 8, 256), 256, 0, stream>>>(head_a, head_b, 14, 8);
  k_conv3x3<<<cdiv(BATCH * D_MODEL * 6 * 6, 256), 256, 0, stream>>>(head_b, c2w, c2b, head_a, 8);
  k_adaptive_pool<<<cdiv(BATCH * D_MODEL * 4 * 4, 256), 256, 0, stream>>>(head_a, head_b, 6, 4);
  k_final_out<<<cdiv(BATCH * 16 * D_MODEL, 256), 256, 0, stream>>>(head_b, out);
}
